// VanillaAttention_19327352832445
// MI455X (gfx1250) — matrix operand order repaired and checked by reading, not compiled, not run on hardware
//
#include <hip/hip_runtime.h>
#include <hip/hip_bf16.h>
#include <stdint.h>

// ---------------------------------------------------------------------------
// Types for CDNA5 WMMA (wave32): v16bf = 16 bf16 (8 VGPRs), v8f = 8 f32.
// ---------------------------------------------------------------------------
typedef __attribute__((ext_vector_type(16))) __bf16 v16bf;
typedef __attribute__((ext_vector_type(8)))  float  v8f;

struct BFrag { uint4 lo, hi; };   // 32 bytes == one v16bf fragment

__device__ __forceinline__ v8f wmma_bf16(const BFrag& a, const BFrag& b, v8f c) {
  v16bf av = __builtin_bit_cast(v16bf, a);
  v16bf bv = __builtin_bit_cast(v16bf, b);
  // (neg_a, A, neg_b, B, c_mod, C, reuse_a, reuse_b)
  return __builtin_amdgcn_wmma_f32_16x16x32_bf16(false, av, false, bv,
                                                 (short)0, c, false, false);
}

__device__ __forceinline__ uint16_t f2bf(float f) {
  uint32_t u = __builtin_bit_cast(uint32_t, f);
  uint32_t r = u + 0x7FFFu + ((u >> 16) & 1u);   // round-to-nearest-even
  return (uint16_t)(r >> 16);
}
__device__ __forceinline__ float bf2f(uint16_t h) {
  uint32_t u = ((uint32_t)h) << 16;
  return __builtin_bit_cast(float, u);
}

// CDNA5 async DMA: copy 16 bytes per lane from global to LDS (ASYNCcnt).
// lds_byte_off = wave-relative LDS byte address (flat LDS addr truncates to it).
__device__ __forceinline__ void async_copy_b128(uint32_t lds_byte_off,
                                                const uint16_t* g) {
  asm volatile("global_load_async_to_lds_b128 %0, %1, off"
               :: "v"(lds_byte_off), "v"(g) : "memory");
}
__device__ __forceinline__ void wait_asynccnt0() {
  asm volatile("s_wait_asynccnt 0" ::: "memory");
}

// A-fragment (16x32 bf16, M x K) from a row-major matrix (ld in elements).
// lane L<16: row M=L, K = {0..7, 16..23}; lane L>=16: row M=L-16, K = {8..15, 24..31}
__device__ __forceinline__ BFrag load_a_frag(const uint16_t* p, int ld) {
  int lane = threadIdx.x & 31;
  const uint16_t* q = p + (size_t)(lane & 15) * ld + ((lane >> 4) << 3);
  BFrag f;
  f.lo = *(const uint4*)(q);        // K = kb .. kb+7
  f.hi = *(const uint4*)(q + 16);   // K = kb+16 .. kb+23
  return f;
}

// B-fragment (32x16 bf16, K x N) from a K-contiguous (i.e. N-major) matrix:
// p points at [n0][k0], ld = elements between consecutive n rows.
// lane L<16: col N=L, K = {0..7, 16..23}; lane L>=16: col N=L-16, K = {8..15, 24..31}
__device__ __forceinline__ BFrag load_bT_frag(const uint16_t* p, int ld) {
  int lane = threadIdx.x & 31;
  const uint16_t* q = p + (size_t)(lane & 15) * ld + ((lane >> 4) << 3);
  BFrag f;
  f.lo = *(const uint4*)(q);        // K = kr .. kr+7
  f.hi = *(const uint4*)(q + 16);   // K = kr+16 .. kr+23
  return f;
}

// ---------------------------------------------------------------------------
// fp32 -> bf16 convert (plain) and convert+transpose (for weights)
// ---------------------------------------------------------------------------
__global__ void cvt_f32_bf16_kernel(const float* __restrict__ in,
                                    uint16_t* __restrict__ out, size_t n) {
  size_t stride = (size_t)gridDim.x * blockDim.x;
  for (size_t i = (size_t)blockIdx.x * blockDim.x + threadIdx.x; i < n; i += stride)
    out[i] = f2bf(in[i]);
}

// in: [Kd, Nd] f32 row-major -> out: [Nd, Kd] bf16 row-major
__global__ void cvtT_f32_bf16_kernel(const float* __restrict__ in,
                                     uint16_t* __restrict__ out, int Kd, int Nd) {
  size_t total = (size_t)Kd * Nd;
  size_t stride = (size_t)gridDim.x * blockDim.x;
  for (size_t idx = (size_t)blockIdx.x * blockDim.x + threadIdx.x; idx < total;
       idx += stride) {
    int k = (int)(idx / Nd), n = (int)(idx % Nd);
    out[(size_t)n * Kd + k] = f2bf(in[idx]);
  }
}

// ---------------------------------------------------------------------------
// Tiled WMMA GEMM: C[M,N] = A[M,K] * Bt[N,K]^T, A/Bt bf16 row-major.
// Block: 256 threads = 8 waves, tile 128x128, K-step 32.
// Double-buffered LDS, tiles staged with GLOBAL_LOAD_ASYNC_TO_LDS_B128.
// Wave w: (w&3) -> 32-row strip, (w>>2) -> 64-col strip; 2x4 grid of 16x16.
// ---------------------------------------------------------------------------
template <bool OUT_BF16>
__global__ __launch_bounds__(256) void gemm_bf16_tn_kernel(
    const uint16_t* __restrict__ A, const uint16_t* __restrict__ Bt,
    void* __restrict__ Cc, int M, int N, int K) {
  __shared__ __align__(16) uint16_t As[2][128 * 40];   // padded stride 40
  __shared__ __align__(16) uint16_t Bs[2][128 * 40];   // Bs[n][k], stride 40

  const int tid  = threadIdx.x;
  const int lane = tid & 31;
  const int wave = tid >> 5;
  const int m0 = blockIdx.y * 128;
  const int n0 = blockIdx.x * 128;
  const int wm = (wave & 3) * 32;
  const int wn = (wave >> 2) * 64;

  const uint32_t asBase = (uint32_t)(uintptr_t)(&As[0][0]);
  const uint32_t bsBase = (uint32_t)(uintptr_t)(&Bs[0][0]);
  const uint32_t bufBytes = 128 * 40 * 2;

  v8f acc[2][4];
#pragma unroll
  for (int i = 0; i < 2; ++i)
#pragma unroll
    for (int j = 0; j < 4; ++j)
      acc[i][j] = (v8f){0.f, 0.f, 0.f, 0.f, 0.f, 0.f, 0.f, 0.f};

  // Stage one 128x32 tile pair into LDS buffer `buf` via async DMA.
  auto stage = [&](int k0, int buf) {
#pragma unroll
    for (int i = 0; i < 2; ++i) {
      int c = tid + i * 256;               // 0..511 chunks of 16B
      int r = c >> 2, off = (c & 3) * 8;   // row, element offset
      uint32_t loff = (uint32_t)(r * 80 + off * 2) + (uint32_t)buf * bufBytes;
      async_copy_b128(asBase + loff, &A [(size_t)(m0 + r) * K + k0 + off]);
      async_copy_b128(bsBase + loff, &Bt[(size_t)(n0 + r) * K + k0 + off]);
    }
  };

  const int nk = K >> 5;
  stage(0, 0);
  for (int kt = 0; kt < nk; ++kt) {
    wait_asynccnt0();      // this wave's DMA into buf[kt&1] done
    __syncthreads();       // everyone's DMA done; prev-buffer readers drained
    if (kt + 1 < nk) stage((kt + 1) << 5, (kt + 1) & 1);  // overlap with WMMA

    const uint16_t* as = &As[kt & 1][0];
    const uint16_t* bs = &Bs[kt & 1][0];
    BFrag af[2], bf[4];
#pragma unroll
    for (int mt = 0; mt < 2; ++mt)
      af[mt] = load_a_frag(as + (wm + mt * 16) * 40, 40);
#pragma unroll
    for (int nt = 0; nt < 4; ++nt)
      bf[nt] = load_bT_frag(bs + (wn + nt * 16) * 40, 40);
#pragma unroll
    for (int mt = 0; mt < 2; ++mt)
#pragma unroll
      for (int nt = 0; nt < 4; ++nt)
        acc[mt][nt] = wmma_bf16(af[mt], bf[nt], acc[mt][nt]);
  }

  // Epilogue. C layout: VGPR r -> row r (lanes 0-15) / row 8+r (lanes 16-31),
  // col = lane & 15.
  const int rowoff = (lane >> 4) * 8;
  const int coll   = lane & 15;
#pragma unroll
  for (int mt = 0; mt < 2; ++mt)
#pragma unroll
    for (int nt = 0; nt < 4; ++nt)
#pragma unroll
      for (int r = 0; r < 8; ++r) {
        int row = m0 + wm + mt * 16 + r + rowoff;
        int col = n0 + wn + nt * 16 + coll;
        float v = acc[mt][nt][r];
        if (OUT_BF16)
          ((uint16_t*)Cc)[(size_t)row * N + col] = f2bf(v);
        else
          ((float*)Cc)[(size_t)row * N + col] = v;
      }
}

// ---------------------------------------------------------------------------
// Repack qkv[B*T, 3C] bf16 -> Q[BH,T,HD] (pre-scaled 1/8), K[BH,T,HD], Vt[BH,HD,T]
// ---------------------------------------------------------------------------
__global__ void repack_qkv_kernel(const uint16_t* __restrict__ qkv,
                                  uint16_t* __restrict__ Qo,
                                  uint16_t* __restrict__ Ko,
                                  uint16_t* __restrict__ Vt) {
  const int T = 2048, C = 1024, HD = 64, NH = 16;
  const size_t total = (size_t)4 * T * 3 * C;
  size_t stride = (size_t)gridDim.x * blockDim.x;
  for (size_t idx = (size_t)blockIdx.x * blockDim.x + threadIdx.x; idx < total;
       idx += stride) {
    size_t row = idx / (3 * C);
    int col = (int)(idx % (3 * C));
    int b = (int)(row / T), t = (int)(row % T);
    int type = col / C, c = col % C;
    int h = c / HD, hd = c % HD;
    size_t bh = (size_t)(b * NH + h);
    uint16_t v = qkv[idx];
    if (type == 0)
      Qo[(bh * T + t) * HD + hd] = f2bf(bf2f(v) * 0.125f);  // 1/sqrt(64)
    else if (type == 1)
      Ko[(bh * T + t) * HD + hd] = v;
    else
      Vt[(bh * HD + hd) * T + t] = v;
  }
}

// ---------------------------------------------------------------------------
// Causal flash attention: grid (T/128, B*NH), 256 threads = 8 waves.
// Each wave owns 16 q rows, streams KV in 32-wide tiles.
// ---------------------------------------------------------------------------
__global__ __launch_bounds__(256) void attn_kernel(
    const uint16_t* __restrict__ Q, const uint16_t* __restrict__ Kk,
    const uint16_t* __restrict__ Vt, uint16_t* __restrict__ Y) {
  const int T = 2048, HD = 64, NH = 16, C = 1024;
  __shared__ __align__(16) uint16_t Pbuf[8][16 * 32];   // per-wave P tile

  const int lane = threadIdx.x & 31;
  const int wave = threadIdx.x >> 5;
  const int bh = blockIdx.y;              // b*NH + h
  const int b = bh >> 4, h = bh & 15;
  const int q0 = blockIdx.x * 128 + wave * 16;

  const uint16_t* Qh = Q  + (size_t)bh * T * HD;
  const uint16_t* Kh = Kk + (size_t)bh * T * HD;
  const uint16_t* Vh = Vt + (size_t)bh * HD * T;

  // Q fragments for this wave's 16 rows (hd halves 0-31, 32-63); reused all tiles.
  BFrag aq[2];
  aq[0] = load_a_frag(Qh + (size_t)q0 * HD,      HD);
  aq[1] = load_a_frag(Qh + (size_t)q0 * HD + 32, HD);

  v8f o[4];
#pragma unroll
  for (int c = 0; c < 4; ++c) o[c] = (v8f){0.f,0.f,0.f,0.f,0.f,0.f,0.f,0.f};
  float m[8], l[8];
#pragma unroll
  for (int r = 0; r < 8; ++r) { m[r] = -1e30f; l[r] = 0.f; }

  const int rowoff = (lane >> 4) * 8;
  const int coll   = lane & 15;
  const int kv_end = q0 + 16;             // causal: last needed kv index is q0+15

  for (int kv0 = 0; kv0 < kv_end; kv0 += 32) {
    // ---- S = Q * K^T over this 32-wide kv tile (two 16x16 C tiles) ----
    v8f s0 = (v8f){0.f,0.f,0.f,0.f,0.f,0.f,0.f,0.f};
    v8f s1 = (v8f){0.f,0.f,0.f,0.f,0.f,0.f,0.f,0.f};
#pragma unroll
    for (int kh = 0; kh < 2; ++kh) {
      BFrag bk0 = load_bT_frag(Kh + (size_t)(kv0)      * HD + kh * 32, HD);
      BFrag bk1 = load_bT_frag(Kh + (size_t)(kv0 + 16) * HD + kh * 32, HD);
      s0 = wmma_bf16(aq[kh], bk0, s0);
      s1 = wmma_bf16(aq[kh], bk1, s1);
    }

    // ---- online softmax (row stats replicated across the 16 lanes) ----
#pragma unroll
    for (int r = 0; r < 8; ++r) {
      int qi = q0 + r + rowoff;
      float v0 = s0[r], v1 = s1[r];
      if (kv0 + coll      > qi) v0 = -1e30f;   // causal mask
      if (kv0 + 16 + coll > qi) v1 = -1e30f;
      float tmax = fmaxf(v0, v1);
#pragma unroll
      for (int off = 8; off > 0; off >>= 1)
        tmax = fmaxf(tmax, __shfl_xor(tmax, off, 32));
      float mn   = fmaxf(m[r], tmax);
      float corr = __expf(m[r] - mn);
      float p0 = __expf(v0 - mn);
      float p1 = __expf(v1 - mn);
      float rs = p0 + p1;
#pragma unroll
      for (int off = 8; off > 0; off >>= 1)
        rs += __shfl_xor(rs, off, 32);
      l[r] = l[r] * corr + rs;
      m[r] = mn;
#pragma unroll
      for (int c = 0; c < 4; ++c) o[c][r] = o[c][r] * corr;

      uint16_t* Pw = &Pbuf[wave][(size_t)(r + rowoff) * 32];
      Pw[coll]      = f2bf(p0);
      Pw[16 + coll] = f2bf(p1);
    }
    asm volatile("s_wait_dscnt 0" ::: "memory");  // P visible before A-frag reload

    // ---- O += P * V (A-frag from LDS, B-frags contiguous from Vt) ----
    BFrag ap = load_a_frag(&Pbuf[wave][0], 32);
#pragma unroll
    for (int c = 0; c < 4; ++c) {
      BFrag bv = load_bT_frag(Vh + (size_t)(c * 16) * T + kv0, T);
      o[c] = wmma_bf16(ap, bv, o[c]);
    }
  }

  // ---- normalize and scatter back to [B,T,C] (heads re-interleaved) ----
#pragma unroll
  for (int r = 0; r < 8; ++r) {
    float linv = 1.0f / l[r];
    int qi = q0 + r + rowoff;
    size_t base = ((size_t)b * T + qi) * C + h * HD;
#pragma unroll
    for (int c = 0; c < 4; ++c)
      Y[base + c * 16 + coll] = f2bf(o[c][r] * linv);
  }
}

// ---------------------------------------------------------------------------
// Host-side launch
// ---------------------------------------------------------------------------
extern "C" void kernel_launch(void* const* d_in, const int* in_sizes, int n_in,
                              void* d_out, int out_size, void* d_ws, size_t ws_size,
                              hipStream_t stream) {
  (void)in_sizes; (void)n_in; (void)out_size; (void)ws_size;
  const float* x      = (const float*)d_in[0];
  const float* w_attn = (const float*)d_in[1];
  const float* w_proj = (const float*)d_in[2];
  float* out = (float*)d_out;

  const size_t B = 4, T = 2048, C = 1024;
  const size_t M = B * T;                 // 8192

  uint8_t* ws = (uint8_t*)d_ws;
  size_t off = 0;
  auto alloc = [&](size_t bytes) -> uint16_t* {
    uint16_t* p = (uint16_t*)(ws + off);
    off += (bytes + 255) & ~(size_t)255;
    return p;
  };
  uint16_t* xb    = alloc(M * C * 2);          // x bf16 [M,K]
  uint16_t* wabT  = alloc(C * 3 * C * 2);      // w_attn bf16 transposed [3C,C]
  uint16_t* wpbT  = alloc(C * C * 2);          // w_proj bf16 transposed [C,C]
  uint16_t* qkvb  = alloc(M * 3 * C * 2);      // qkv bf16
  uint16_t* Qb    = alloc(M * C * 2);          // [BH,T,HD] scaled
  uint16_t* Kb    = alloc(M * C * 2);          // [BH,T,HD]
  uint16_t* Vtb   = alloc(M * C * 2);          // [BH,HD,T]
  uint16_t* Yb    = alloc(M * C * 2);          // attn output bf16 [B,T,C]

  cvt_f32_bf16_kernel <<<2048, 256, 0, stream>>>(x, xb, M * C);
  cvtT_f32_bf16_kernel<<<2048, 256, 0, stream>>>(w_attn, wabT, (int)C, (int)(3 * C));
  cvtT_f32_bf16_kernel<<<1024, 256, 0, stream>>>(w_proj, wpbT, (int)C, (int)C);

  // qkv = x @ w_attn : [8192,1024] x [1024,3072] (B pre-transposed)
  gemm_bf16_tn_kernel<true><<<dim3(24, 64), 256, 0, stream>>>(
      xb, wabT, (void*)qkvb, (int)M, (int)(3 * C), (int)C);

  repack_qkv_kernel<<<4096, 256, 0, stream>>>(qkvb, Qb, Kb, Vtb);

  // causal flash attention: (T/128 q-tiles, B*NH heads)
  attn_kernel<<<dim3(16, 64), 256, 0, stream>>>(Qb, Kb, Vtb, Yb);

  // out = y @ w_proj : [8192,1024] x [1024,1024], fp32 out
  gemm_bf16_tn_kernel<false><<<dim3(8, 64), 256, 0, stream>>>(
      Yb, wpbT, (void*)out, (int)M, (int)C, (int)C);
}
